// TDNet_32882269618774
// MI455X (gfx1250) — matrix operand order repaired and checked by reading, not compiled, not run on hardware
//
#include <hip/hip_runtime.h>
#include <hip/hip_bf16.h>
#include <cstdint>
#include <cstddef>

// ---------------------------------------------------------------------------
// Types / helpers
// ---------------------------------------------------------------------------
typedef unsigned short u16;
typedef uint32_t u32;
typedef uint64_t u64;
typedef __attribute__((ext_vector_type(4)))  u32    u32x4;
typedef __attribute__((ext_vector_type(4)))  int    i32x4;
typedef __attribute__((ext_vector_type(8)))  int    i32x8;
typedef __attribute__((ext_vector_type(16))) __bf16 v16bf;
typedef __attribute__((ext_vector_type(8)))  float  v8f;

#define GLOBAL_AS __attribute__((address_space(1)))
#define LDS_AS    __attribute__((address_space(3)))

union FragBf { u32x4 q[2]; v16bf v; };

__device__ __forceinline__ u16 f2bf(float f) {
  union { float f; u32 u; } v; v.f = f;
  u32 u = v.u;
  u32 r = (u + 0x7FFFu + ((u >> 16) & 1u)) >> 16;   // round-to-nearest-even
  return (u16)r;
}
__device__ __forceinline__ float bf2f(u16 h) {
  union { u32 u; float f; } v; v.u = ((u32)h) << 16; return v.f;
}
__device__ __forceinline__ float sigmoidf(float x) { return 1.0f / (1.0f + __expf(-x)); }

// ---- CDNA5 async global->LDS copy (ASYNCcnt) with scalar fallback ----------
#if __has_builtin(__builtin_amdgcn_global_load_async_to_lds_b128) && \
    __has_builtin(__builtin_amdgcn_s_wait_asynccnt)
#define HAVE_ASYNC_LDS 1
#else
#define HAVE_ASYNC_LDS 0
#endif

__device__ __forceinline__ void copy16_to_lds(const u16* gsrc, u16* ldst) {
#if HAVE_ASYNC_LDS
  // AS1 pointer value == generic address; AS3 pointers are 32-bit and the low
  // 32 bits of a generic LDS address are the LDS byte offset, so inttoptr
  // truncation yields the correct LDS address (same pattern CK uses).
  __builtin_amdgcn_global_load_async_to_lds_b128(
      (GLOBAL_AS i32x4*)(uintptr_t)gsrc,
      (LDS_AS i32x4*)(uintptr_t)ldst,
      0, 0);
#else
  *(u32x4*)ldst = *(const u32x4*)gsrc;
#endif
}
__device__ __forceinline__ void async_wait0() {
#if HAVE_ASYNC_LDS
  __builtin_amdgcn_s_wait_asynccnt(0);
#endif
}

// ---- CDNA5 Tensor Data Mover (TENSORcnt) with scalar fallback --------------
#if __has_builtin(__builtin_amdgcn_tensor_load_to_lds) && \
    __has_builtin(__builtin_amdgcn_s_wait_tensorcnt)
#define HAVE_TDM 1
#else
#define HAVE_TDM 0
#endif

#if HAVE_TDM
// 2-D bf16 tile load: tile_k elems per row (contiguous), `rows` rows,
// global row stride `stride_elems`; packed densely into LDS at lds_byte_addr.
__device__ __forceinline__ void tdm_load_tile_bf16(u32 lds_byte_addr, const void* gaddr,
                                                   u32 tile_k, u32 rows, u32 stride_elems)
{
  u64 ga = (u64)(uintptr_t)gaddr;
  u32x4 g0;
  g0.x = 1u;                                            // count=1, user descriptor
  g0.y = lds_byte_addr;                                 // lds_addr
  g0.z = (u32)ga;                                       // global_addr[31:0]
  g0.w = ((u32)(ga >> 32) & 0x01FFFFFFu) | (2u << 30);  // global_addr[56:32] | type=2
  i32x8 g1;
  g1[0] = (1 << 16);                                    // data_size=1 (2 bytes)
  g1[1] = (int)((tile_k & 0xFFFFu) << 16);              // tensor_dim0[15:0]
  g1[2] = (int)(((tile_k >> 16) & 0xFFFFu) | ((rows & 0xFFFFu) << 16));   // td0 hi | td1 lo
  g1[3] = (int)(((rows >> 16) & 0xFFFFu) | ((tile_k & 0xFFFFu) << 16));   // td1 hi | tile_dim0
  g1[4] = (int)(rows & 0xFFFFu);                        // tile_dim1 | tile_dim2=0
  g1[5] = (int)stride_elems;                            // tensor_dim0_stride[31:0]
  g1[6] = 0;                                            // stride0 hi | dim1_stride lo
  g1[7] = 0;
  i32x4 z4 = {0, 0, 0, 0};
#if defined(__clang_major__) && (__clang_major__ >= 23)
  i32x8 z8 = {0, 0, 0, 0, 0, 0, 0, 0};
  __builtin_amdgcn_tensor_load_to_lds(g0, g1, z4, z4, z8, 0);
#else
  __builtin_amdgcn_tensor_load_to_lds(g0, g1, z4, z4, 0);
#endif
}
#endif

// A-matrix (16x32 bf16) fragment: lane<16 -> row=lane, K {0..7,16..23};
// lane>=16 -> row=lane-16, K {8..15,24..31}.  src: row-major, stride `ld` u16.
__device__ __forceinline__ v16bf load_a_frag(const u16* p, int ld, int lane) {
  int l = lane & 15, hi = lane >> 4;
  FragBf f;
  f.q[0] = *(const u32x4*)(p + (size_t)l * ld + hi * 8);
  f.q[1] = *(const u32x4*)(p + (size_t)l * ld + 16 + hi * 8);
  return f.v;
}
// B-matrix (32x16 bf16) fragment, operand stored N-major (rows = N, K contiguous):
// lane<16 -> col=lane, K {0..7,16..23}; lane>=16 -> col=lane-16, K {8..15,24..31}.
__device__ __forceinline__ v16bf load_b_frag(const u16* p, int ld, int lane) {
  int l = lane & 15, hi = lane >> 4;
  FragBf f;
  f.q[0] = *(const u32x4*)(p + (size_t)l * ld + hi * 8);
  f.q[1] = *(const u32x4*)(p + (size_t)l * ld + 16 + hi * 8);
  return f.v;
}
__device__ __forceinline__ v8f wmma_bf16(v16bf a, v16bf b, v8f c) {
  return __builtin_amdgcn_wmma_f32_16x16x32_bf16(false, a, false, b, (short)0, c, false, false);
}

// ---------------------------------------------------------------------------
// Generic WMMA GEMM: C[M,N] = A[M,K](bf16) * op(B) (+bias),  M,N %64==0, K%32==0
//   B_IS_NK = true : B is row-major N x K  (C = A * B^T)
//   B_IS_NK = false: B is row-major K x N  (C = A * B, transpose on stage)
// Block: 256 thr (8 waves), tile 64x64, waves 4x2, each wave 16x32.
// LDS double-buffered; next tile staged with async global->LDS copies.
// ---------------------------------------------------------------------------
template <bool B_IS_NK>
__global__ __launch_bounds__(256)
void gemm_bf16_kernel(const u16* __restrict__ A, const u16* __restrict__ Bm,
                      const float* __restrict__ bias,
                      float* __restrict__ Cf, u16* __restrict__ Cbf,
                      int M, int N, int K)
{
  __shared__ u16 As[2][64 * 32];
  __shared__ u16 Bs[2][64 * 32];
  const int tid  = threadIdx.x;
  const int lane = tid & 31, wave = tid >> 5;
  const int n0 = blockIdx.x * 64, m0 = blockIdx.y * 64;
  const int wm = wave >> 1, wn = wave & 1;

  v8f acc0 = {}, acc1 = {};
  const int arow = tid >> 2, achunk = (tid & 3) * 8;

  auto stage = [&](int k0, int buf) {
    copy16_to_lds(&A[(size_t)(m0 + arow) * K + k0 + achunk], &As[buf][arow * 32 + achunk]);
    if (B_IS_NK) {
      copy16_to_lds(&Bm[(size_t)(n0 + arow) * K + k0 + achunk], &Bs[buf][arow * 32 + achunk]);
    } else {
      for (int idx = tid; idx < 64 * 32; idx += 256) {
        int n = idx >> 5, k = idx & 31;
        Bs[buf][n * 32 + k] = Bm[(size_t)(k0 + k) * N + n0 + n];
      }
    }
  };

  stage(0, 0);
  async_wait0();
  __syncthreads();

  const int nk = K >> 5;
  for (int kk = 0; kk < nk; ++kk) {
    const int buf = kk & 1;
    if (kk + 1 < nk) {
      stage((kk + 1) * 32, buf ^ 1);
      __builtin_prefetch(&A[(size_t)(m0 + arow) * K + (kk + 1) * 32 + achunk], 0, 0);
    }
    v16bf af = load_a_frag(&As[buf][(wm * 16) * 32], 32, lane);
    v16bf b0 = load_b_frag(&Bs[buf][(wn * 32) * 32], 32, lane);
    v16bf b1 = load_b_frag(&Bs[buf][(wn * 32 + 16) * 32], 32, lane);
    acc0 = wmma_bf16(af, b0, acc0);
    acc1 = wmma_bf16(af, b1, acc1);
    async_wait0();
    __syncthreads();
  }

  const int mrow  = m0 + wm * 16 + (lane >> 4) * 8;
  const int ncol0 = n0 + wn * 32 + (lane & 15);
#pragma unroll
  for (int i = 0; i < 8; ++i) {
    float v0 = acc0[i], v1 = acc1[i];
    if (bias) { v0 += bias[ncol0]; v1 += bias[ncol0 + 16]; }
    size_t m = (size_t)(mrow + i);
    if (Cf)  { Cf[m * N + ncol0] = v0;        Cf[m * N + ncol0 + 16] = v1; }
    if (Cbf) { Cbf[m * N + ncol0] = f2bf(v0); Cbf[m * N + ncol0 + 16] = f2bf(v1); }
  }
}

// ---------------------------------------------------------------------------
// Fused GRU step: g = h_in @ Whh^T + bhh ; gates ; h_out, seq(bf16) out.
// Scan length = 64 (reference scans over the batch axis).  Block = 128 thr
// (4 waves), tile: 16 rows (inner batch) x 64 gate-cols x 3 gates.
// grid = (768/64 strips, M/16, 2 dirs).  dir==1 processes s = 63-step.
// Gate-weight tiles fetched by the Tensor Data Mover when available.
// ---------------------------------------------------------------------------
__global__ __launch_bounds__(128)
void gru_step_kernel(const float* __restrict__ h_in,   // [2][M][768]
                     float* __restrict__ h_out,        // [2][M][768]
                     const u16* __restrict__ xp_f,     // [64*M][2304] bf16 (x@Wih^T+bih)
                     const u16* __restrict__ xp_b,
                     const u16* __restrict__ Whh_f,    // [2304][768] bf16
                     const u16* __restrict__ Whh_b,
                     const float* __restrict__ bhh_f,
                     const float* __restrict__ bhh_b,
                     u16* __restrict__ seq_out,        // [64][M][1536] bf16 (f|b concat)
                     int step, int M)
{
  __shared__ u16   hS[16 * 32];
  __shared__ u16   wS[3][64 * 32];
  __shared__ float gS[3][16 * 64];

  const int tid = threadIdx.x, lane = tid & 31, wave = tid >> 5;
  const int strip = blockIdx.x;       // 0..11
  const int mt    = blockIdx.y;       // 0..M/16-1
  const int dir   = blockIdx.z;       // 0 fwd, 1 bwd
  const int s     = dir ? (63 - step) : step;
  const u16*   xp  = dir ? xp_b  : xp_f;
  const u16*   Whh = dir ? Whh_b : Whh_f;
  const float* bhh = dir ? bhh_b : bhh_f;
  const float* hin = h_in  + (size_t)dir * M * 768;
  float*       hout = h_out + (size_t)dir * M * 768;
  const int n0 = strip * 64;

  v8f acc[3] = {};

  for (int k0 = 0; k0 < 768; k0 += 32) {
    __syncthreads();
    // stage h tile (f32 -> bf16 convert, all threads)
    for (int idx = tid; idx < 16 * 32; idx += 128) {
      int p = idx >> 5, k = idx & 31;
      hS[idx] = f2bf(hin[(size_t)(mt * 16 + p) * 768 + k0 + k]);
    }
#if HAVE_TDM
    if (tid == 0) {
      u32 lds0 = (u32)(uintptr_t)&wS[0][0];   // low 32 bits == LDS byte offset
#pragma unroll
      for (int g = 0; g < 3; ++g)
        tdm_load_tile_bf16(lds0 + (u32)g * 64 * 32 * 2,
                           &Whh[(size_t)(g * 768 + n0) * 768 + k0],
                           /*tile_k=*/32, /*rows=*/64, /*stride=*/768);
      __builtin_amdgcn_s_wait_tensorcnt(0);
    }
#else
    for (int idx = tid; idx < 3 * 64 * 4; idx += 128) {
      int g = idx >> 8, rem = idx & 255;
      int r = rem >> 2, chunk = (rem & 3) * 8;
      *(u32x4*)&wS[g][r * 32 + chunk] =
          *(const u32x4*)&Whh[(size_t)(g * 768 + n0 + r) * 768 + k0 + chunk];
    }
#endif
    __syncthreads();
    v16bf af = load_a_frag(hS, 32, lane);
#pragma unroll
    for (int g = 0; g < 3; ++g) {
      v16bf bf = load_b_frag(&wS[g][(wave * 16) * 32], 32, lane);
      acc[g] = wmma_bf16(af, bf, acc[g]);
    }
  }
  __syncthreads();

  // spill g tiles (+bhh) to LDS for the elementwise gate phase
  {
    const int mr = (lane >> 4) * 8;
    const int nc = wave * 16 + (lane & 15);
#pragma unroll
    for (int g = 0; g < 3; ++g) {
      float b = bhh[g * 768 + n0 + nc];
#pragma unroll
      for (int i = 0; i < 8; ++i)
        gS[g][(mr + i) * 64 + nc] = acc[g][i] + b;
    }
  }
  __syncthreads();

  for (int idx = tid; idx < 16 * 64; idx += 128) {
    int p = idx >> 6, j = idx & 63;
    int col  = n0 + j;
    int prow = mt * 16 + p;
    size_t xrow = ((size_t)s * M + prow) * 2304;
    float xr = bf2f(xp[xrow + col]);
    float xz = bf2f(xp[xrow + 768 + col]);
    float xn = bf2f(xp[xrow + 1536 + col]);
    float r = sigmoidf(xr + gS[0][p * 64 + j]);
    float z = sigmoidf(xz + gS[1][p * 64 + j]);
    float n = tanhf(xn + r * gS[2][p * 64 + j]);
    float hp = hin[(size_t)prow * 768 + col];
    float hn = (1.0f - z) * n + z * hp;
    hout[(size_t)prow * 768 + col] = hn;
    seq_out[((size_t)s * M + prow) * 1536 + dir * 768 + col] = f2bf(hn);
  }
}

// ---------------------------------------------------------------------------
// S[b] = tpW[b](16x1536) * tw[b]^T(1536x128).  One block per b, 8 waves,
// wave w -> 16 columns of l.  Fragments straight from global (K contiguous).
// ---------------------------------------------------------------------------
__global__ __launch_bounds__(256)
void attn_s_kernel(const u16* __restrict__ tpW,   // [64][16][1536] bf16
                   const u16* __restrict__ tw,    // [64][128][1536] bf16
                   float* __restrict__ S)         // [64][16][128]
{
  const int b = blockIdx.x;
  const int tid = threadIdx.x, lane = tid & 31, wave = tid >> 5;
  const u16* A  = tpW + (size_t)b * 16 * 1536;
  const u16* Bp = tw  + (size_t)b * 128 * 1536 + (size_t)(wave * 16) * 1536;
  v8f acc = {};
  for (int k0 = 0; k0 < 1536; k0 += 32) {
    v16bf af = load_a_frag(A + k0, 1536, lane);
    v16bf bf = load_b_frag(Bp + k0, 1536, lane);
    acc = wmma_bf16(af, bf, acc);
  }
  const int mr = (lane >> 4) * 8, nc = wave * 16 + (lane & 15);
#pragma unroll
  for (int i = 0; i < 8; ++i)
    S[(size_t)b * 16 * 128 + (size_t)(mr + i) * 128 + nc] = acc[i];
}

// One wave per (b,t) row of 128: softmax over l, emit bf16 attention.
__global__ __launch_bounds__(256)
void softmax_kernel(const float* __restrict__ S, u16* __restrict__ attbf)
{
  const int tid = threadIdx.x, lane = tid & 31, wave = tid >> 5;
  const int row = blockIdx.x * 8 + wave;           // 0..1023
  const float* p = S + (size_t)row * 128;
  float v[4];
#pragma unroll
  for (int i = 0; i < 4; ++i) v[i] = p[lane * 4 + i];
  float m = fmaxf(fmaxf(v[0], v[1]), fmaxf(v[2], v[3]));
  for (int o = 16; o > 0; o >>= 1) m = fmaxf(m, __shfl_xor(m, o, 32));
  float s = 0.f;
#pragma unroll
  for (int i = 0; i < 4; ++i) { v[i] = __expf(v[i] - m); s += v[i]; }
  for (int o = 16; o > 0; o >>= 1) s += __shfl_xor(s, o, 32);
  const float inv = 1.0f / s;
  u16* q = attbf + (size_t)row * 128;
#pragma unroll
  for (int i = 0; i < 4; ++i) q[lane * 4 + i] = f2bf(v[i] * inv);
}

// r[b] = att[b](16x128) * tw[b](128x1536).  grid=(12 strips of 128, 64 b).
__global__ __launch_bounds__(256)
void attn_r_kernel(const u16* __restrict__ attbf,  // [64][16][128] bf16
                   const u16* __restrict__ tw,     // [64][128][1536] bf16
                   float* __restrict__ r)          // [64][16][1536]
{
  __shared__ u16 aS[16 * 32];
  __shared__ u16 bS[128 * 32];
  const int strip = blockIdx.x, b = blockIdx.y;
  const int tid = threadIdx.x, lane = tid & 31, wave = tid >> 5;
  const int n0 = strip * 128;
  v8f acc = {};
  for (int k0 = 0; k0 < 128; k0 += 32) {
    __syncthreads();
    for (int idx = tid; idx < 16 * 32; idx += 256) {
      int mm = idx >> 5, k = idx & 31;
      aS[idx] = attbf[(size_t)b * 2048 + (size_t)mm * 128 + k0 + k];
    }
    for (int idx = tid; idx < 128 * 32; idx += 256) {   // transpose tw tile
      int n = idx >> 5, k = idx & 31;
      bS[idx] = tw[(size_t)b * 128 * 1536 + (size_t)(k0 + k) * 1536 + n0 + n];
    }
    __syncthreads();
    v16bf af = load_a_frag(aS, 32, lane);
    v16bf bf = load_b_frag(&bS[(wave * 16) * 32], 32, lane);
    acc = wmma_bf16(af, bf, acc);
  }
  const int mr = (lane >> 4) * 8, nc = n0 + wave * 16 + (lane & 15);
#pragma unroll
  for (int i = 0; i < 8; ++i)
    r[(size_t)b * 16 * 1536 + (size_t)(mr + i) * 1536 + nc] = acc[i];
}

// R[b] = concat(r[b,15,:], sum_{t<15} beta[t]*r[b,t,:])
__global__ void meanR_kernel(const float* __restrict__ r,
                             const float* __restrict__ beta,
                             float* __restrict__ R)
{
  int idx = blockIdx.x * blockDim.x + threadIdx.x;   // 64*1536
  if (idx >= 64 * 1536) return;
  int b = idx / 1536, d = idx - b * 1536;
  const float* rb = r + (size_t)b * 16 * 1536;
  R[(size_t)b * 3072 + d] = rb[(size_t)15 * 1536 + d];
  float a = 0.f;
#pragma unroll
  for (int t = 0; t < 15; ++t) a += beta[t] * rb[(size_t)t * 1536 + d];
  R[(size_t)b * 3072 + 1536 + d] = a;
}

// out = R @ det_W^T + det_b : one wave per (b,o) output.
__global__ __launch_bounds__(128)
void head_kernel(const float* __restrict__ R, const float* __restrict__ detW,
                 const float* __restrict__ detb, float* __restrict__ out)
{
  const int tid = threadIdx.x, lane = tid & 31, wave = tid >> 5;
  const int g = blockIdx.x * 4 + wave;   // 0..127
  const int b = g >> 1, o = g & 1;
  const float* Rb = R + (size_t)b * 3072;
  const float* W  = detW + (size_t)o * 3072;
  float s = 0.f;
  for (int k = lane; k < 3072; k += 32) s += Rb[k] * W[k];
  for (int off = 16; off > 0; off >>= 1) s += __shfl_xor(s, off, 32);
  if (lane == 0) out[b * 2 + o] = s + detb[o];
}

// ---- small utility kernels -------------------------------------------------
__global__ void gather_kernel(const int* __restrict__ ids,
                              const float* __restrict__ emb,
                              u16* __restrict__ outp, int nrows)
{
  size_t i = (size_t)blockIdx.x * blockDim.x + threadIdx.x;
  size_t total = (size_t)nrows * 768;
  if (i >= total) return;
  size_t row = i / 768, h = i - row * 768;
  outp[i] = f2bf(emb[(size_t)ids[row] * 768 + h]);
}
__global__ void f2bf_kernel(const float* __restrict__ src, u16* __restrict__ dst, size_t n)
{
  size_t i = (size_t)blockIdx.x * blockDim.x + threadIdx.x;
  if (i < n) dst[i] = f2bf(src[i]);
}
__global__ void zero_kernel(float* __restrict__ p, size_t n)
{
  size_t i = (size_t)blockIdx.x * blockDim.x + threadIdx.x;
  if (i < n) p[i] = 0.f;
}

// ---------------------------------------------------------------------------
// Host orchestration
// ---------------------------------------------------------------------------
extern "C" void kernel_launch(void* const* d_in, const int* in_sizes, int n_in,
                              void* d_out, int out_size, void* d_ws, size_t ws_size,
                              hipStream_t stream)
{
  (void)in_sizes; (void)n_in; (void)out_size; (void)ws_size;

  const int*   tweet = (const int*)  d_in[0];
  const int*   topic = (const int*)  d_in[1];
  const float* beta  = (const float*)d_in[2];
  const float* emb   = (const float*)d_in[3];
  const float* Wih_f = (const float*)d_in[4];
  const float* Whh_f = (const float*)d_in[5];
  const float* bih_f = (const float*)d_in[6];
  const float* bhh_f = (const float*)d_in[7];
  const float* Wih_b = (const float*)d_in[8];
  const float* Whh_b = (const float*)d_in[9];
  const float* bih_b = (const float*)d_in[10];
  const float* bhh_b = (const float*)d_in[11];
  const float* W_bl  = (const float*)d_in[12];
  const float* det_W = (const float*)d_in[13];
  const float* det_b = (const float*)d_in[14];
  float* outp = (float*)d_out;       // [64*2]
  float* R    = outp + 128;          // [64*3072]

  // --- workspace layout (256B-aligned slabs) ---
  char* ws = (char*)d_ws;
  size_t off = 0;
  auto alloc = [&](size_t bytes) -> char* {
    char* p = ws + off;
    off += (bytes + 255) & ~(size_t)255;
    return p;
  };
  u16* xtw    = (u16*)alloc((size_t)8192 * 768 * 2);
  u16* xtp    = (u16*)alloc((size_t)1024 * 768 * 2);
  u16* wihf   = (u16*)alloc((size_t)2304 * 768 * 2);
  u16* whhf   = (u16*)alloc((size_t)2304 * 768 * 2);
  u16* wihb   = (u16*)alloc((size_t)2304 * 768 * 2);
  u16* whhb   = (u16*)alloc((size_t)2304 * 768 * 2);
  u16* wbl    = (u16*)alloc((size_t)1536 * 1536 * 2);
  u16* xp_tw_f= (u16*)alloc((size_t)8192 * 2304 * 2);
  u16* xp_tw_b= (u16*)alloc((size_t)8192 * 2304 * 2);
  u16* xp_tp_f= (u16*)alloc((size_t)1024 * 2304 * 2);
  u16* xp_tp_b= (u16*)alloc((size_t)1024 * 2304 * 2);
  u16* twbf   = (u16*)alloc((size_t)64 * 128 * 1536 * 2);
  u16* tpbf   = (u16*)alloc((size_t)64 * 16 * 1536 * 2);
  u16* tpWbf  = (u16*)alloc((size_t)1024 * 1536 * 2);
  u16* attbf  = (u16*)alloc((size_t)1024 * 128 * 2);
  float* hA   = (float*)alloc((size_t)2 * 128 * 768 * 4);
  float* hB   = (float*)alloc((size_t)2 * 128 * 768 * 4);
  float* Sbuf = (float*)alloc((size_t)1024 * 128 * 4);
  float* rbuf = (float*)alloc((size_t)64 * 16 * 1536 * 4);

  const size_t nW  = (size_t)2304 * 768;
  const size_t nBl = (size_t)1536 * 1536;

  // 1) weight converts to bf16
  f2bf_kernel<<<(int)((nW + 255) / 256), 256, 0, stream>>>(Wih_f, wihf, nW);
  f2bf_kernel<<<(int)((nW + 255) / 256), 256, 0, stream>>>(Whh_f, whhf, nW);
  f2bf_kernel<<<(int)((nW + 255) / 256), 256, 0, stream>>>(Wih_b, wihb, nW);
  f2bf_kernel<<<(int)((nW + 255) / 256), 256, 0, stream>>>(Whh_b, whhb, nW);
  f2bf_kernel<<<(int)((nBl + 255) / 256), 256, 0, stream>>>(W_bl, wbl, nBl);

  // 2) embedding gathers (f32 -> bf16)
  gather_kernel<<<(int)(((size_t)8192 * 768 + 255) / 256), 256, 0, stream>>>(tweet, emb, xtw, 8192);
  gather_kernel<<<(int)(((size_t)1024 * 768 + 255) / 256), 256, 0, stream>>>(topic, emb, xtp, 1024);

  // 3) input projections xp = x @ Wih^T + bih  (bf16 out)
  gemm_bf16_kernel<true><<<dim3(36, 128), 256, 0, stream>>>(xtw, wihf, bih_f, nullptr, xp_tw_f, 8192, 2304, 768);
  gemm_bf16_kernel<true><<<dim3(36, 128), 256, 0, stream>>>(xtw, wihb, bih_b, nullptr, xp_tw_b, 8192, 2304, 768);
  gemm_bf16_kernel<true><<<dim3(36,  16), 256, 0, stream>>>(xtp, wihf, bih_f, nullptr, xp_tp_f, 1024, 2304, 768);
  gemm_bf16_kernel<true><<<dim3(36,  16), 256, 0, stream>>>(xtp, wihb, bih_b, nullptr, xp_tp_b, 1024, 2304, 768);

  // 4) tweet bi-GRU: 64 sequential steps (M = 128), h ping-pong
  zero_kernel<<<(int)(((size_t)2 * 128 * 768 + 255) / 256), 256, 0, stream>>>(hA, (size_t)2 * 128 * 768);
  for (int k = 0; k < 64; ++k) {
    const float* hi = (k & 1) ? hB : hA;
    float*       ho = (k & 1) ? hA : hB;
    gru_step_kernel<<<dim3(12, 8, 2), 128, 0, stream>>>(
        hi, ho, xp_tw_f, xp_tw_b, whhf, whhb, bhh_f, bhh_b, twbf, k, 128);
  }

  // 5) topic bi-GRU: 64 sequential steps (M = 16)
  zero_kernel<<<(int)(((size_t)2 * 16 * 768 + 255) / 256), 256, 0, stream>>>(hA, (size_t)2 * 16 * 768);
  for (int k = 0; k < 64; ++k) {
    const float* hi = (k & 1) ? hB : hA;
    float*       ho = (k & 1) ? hA : hB;
    gru_step_kernel<<<dim3(12, 1, 2), 128, 0, stream>>>(
        hi, ho, xp_tp_f, xp_tp_b, whhf, whhb, bhh_f, bhh_b, tpbf, k, 16);
  }

  // 6) tpW = tp @ W_bl   (C = A*B, bf16 out)
  gemm_bf16_kernel<false><<<dim3(24, 16), 256, 0, stream>>>(tpbf, wbl, nullptr, nullptr, tpWbf, 1024, 1536, 1536);

  // 7) attention scores, softmax, context
  attn_s_kernel<<<64, 256, 0, stream>>>(tpWbf, twbf, Sbuf);
  softmax_kernel<<<128, 256, 0, stream>>>(Sbuf, attbf);
  attn_r_kernel<<<dim3(12, 64), 256, 0, stream>>>(attbf, twbf, rbuf);

  // 8) pooling into R, then the 2-way head
  meanR_kernel<<<(64 * 1536 + 255) / 256, 256, 0, stream>>>(rbuf, beta, R);
  head_kernel<<<32, 128, 0, stream>>>(R, det_W, det_b, outp);
}
